// GATv2LSTM_49864570306627
// MI455X (gfx1250) — compile-verified, matrix operand order checked by reading, not run on hardware
//
#include <hip/hip_runtime.h>
#include <hip/hip_bf16.h>
#include <math.h>

typedef _Float16 half_t;
typedef __attribute__((ext_vector_type(16))) _Float16 v16h;
typedef __attribute__((ext_vector_type(8)))  _Float16 v8h;
typedef __attribute__((ext_vector_type(8)))  float    v8f;
typedef __attribute__((ext_vector_type(4)))  unsigned int v4u;
typedef __attribute__((ext_vector_type(8)))  int v8i;
typedef __attribute__((ext_vector_type(4)))  int v4i;

#define DEVFN __device__ __forceinline__

#if __has_builtin(__builtin_amdgcn_tensor_load_to_lds)
#define HAVE_TDM 1
#else
#define HAVE_TDM 0
#endif

// ---------------- problem constants ----------------
constexpr int  Bc = 2, T_INc = 12, Nn = 2000, Ee = 32000, GHc = 32, LHc = 128, T_OUTc = 12;
constexpr int  NG = Bc * T_INc;          // 24 graphs
constexpr int  ET = Ee + Nn;             // 34000 edges (incl self loops)
constexpr long NN = (long)NG * Nn;       // 48000 node-rows
constexpr int  BNr = Bc * Nn;            // 4000 LSTM batch
constexpr float NEG_SLOPE = 0.2f, EPSc = 1e-5f;

// ---------------- workspace arena (byte offsets, lifetime-aliased) ----------------
constexpr size_t SZROW = (size_t)NN * 512 * 4;        // 98,304,000
constexpr size_t OFF_A = 0;                            // XL1  -> GATESX
constexpr size_t OFF_B = SZROW;                        // XR1  -> S1H/INH + SEQOUT
constexpr size_t OFF_C = 2 * SZROW;                    // OUT1 -> gat2 bufs -> head bufs
constexpr size_t OFF_D = 3 * SZROW;                    // LOG1/M1/DEN1 -> lstm tmp; weights
constexpr size_t D_LOG1 = OFF_D;                       // 13,056,000
constexpr size_t D_M1   = OFF_D + 13056000;            // 768,000 (uint keys)
constexpr size_t D_DEN1 = OFF_D + 13824000;            // 768,000
constexpr size_t D_W    = OFF_D + 14592000;            // persistent f16 weights (padded)
constexpr size_t D_W1H  = D_W;                         // 256x128        -> 65,536
constexpr size_t D_W2H  = D_W + 65536;                 // 128x64 padded  -> 16,384
constexpr size_t D_WL2H = D_W + 81920;                 // 512x64 padded  -> 65,536
constexpr size_t D_WR2H = D_W + 147456;                // 512x64 padded  -> 65,536
constexpr size_t D_WIHT = D_W + 212992;                // 4 x 262,144
constexpr size_t D_WHHT = D_W + 1261568;               // 4 x 131,072
constexpr size_t D_BL2P = D_W + 1785856;               // 64 f32 padded bias
constexpr size_t D_BR2P = D_W + 1786112;
constexpr size_t D_B2P  = D_W + 1786368;
// LSTM temporaries overlay LOG1/M1 (dead after GAT1):
constexpr size_t D_GATESH = OFF_D;                     // 8,192,000
constexpr size_t D_HH     = OFF_D + 8192000;           // 1,024,000 (f16)
constexpr size_t D_CBUF   = OFF_D + 9216000;           // 2,048,000
constexpr size_t D_CTXH   = OFF_D + 11264000;          // 2,048,000 (f16)
// region C overlays (OUT1 dead after fin1); XL2/XR2 now stride-64:
constexpr size_t C_XL2  = OFF_C;                       // 12,288,000
constexpr size_t C_XR2  = OFF_C + 12288000;            // 12,288,000
constexpr size_t C_OUT2 = OFF_C + 24576000;            //  6,144,000 (stride 32)
constexpr size_t C_LOG2 = OFF_C + 30720000;            //  3,264,000
constexpr size_t C_M2   = OFF_C + 33984000;
constexpr size_t C_DEN2 = OFF_C + 34176000;
// head phase (all C dead by then):
constexpr size_t C_H1   = OFF_C;
constexpr size_t C_H1H  = OFF_C + 2048000;
constexpr size_t C_OUTH = OFF_C + 3072000;             // 4000x64 f32
// region B overlays:
constexpr size_t B_S1H  = OFF_B;                        // also INH (f16)
constexpr size_t B_SEQ  = OFF_B + 49152000;             // f32 LSTM layer output

// ---------------- helpers ----------------
DEVFN float sigm(float x) { return 1.0f / (1.0f + expf(-x)); }
DEVFN unsigned fkey(float f) {
  unsigned u = __float_as_uint(f);
  return (u & 0x80000000u) ? ~u : (u | 0x80000000u);
}
DEVFN float fromkey(unsigned k) {
  unsigned u = (k & 0x80000000u) ? (k & 0x7fffffffu) : ~k;
  return __uint_as_float(u);
}
DEVFN void edge_nodes(const int* ei, int e, int& s, int& d) {
  if (e < Ee) { s = ei[e]; d = ei[Ee + e]; } else { s = e - Ee; d = e - Ee; }
}

#if HAVE_TDM
// Issue a TDM 2D tile load: 32 rows x 32 f16, row stride = lda elements, into LDS.
// D# layout per CDNA5 ISA ch.8 (group0: count/lds_addr/global_addr/type=2;
// group1: data_size=2B, tensor dims huge (tiles always in-bounds), tile 32x32, stride0=lda).
DEVFN void tdm_load_a(const half_t* src, half_t* dstLds, int lda) {
  unsigned lds_a = (unsigned)(uintptr_t)(void*)dstLds;   // LDS_ADDR = flat[31:0]
  unsigned long long ga = (unsigned long long)(uintptr_t)src;
  v4u g0 = { 1u, lds_a, (unsigned)(ga & 0xffffffffu),
             (unsigned)((ga >> 32) & 0x01ffffffu) | (2u << 30) };
  unsigned td0 = 0x7fffffffu, td1 = 0x7fffffffu;
  unsigned long long st0 = (unsigned long long)lda;
  v8i g1 = { (int)(1u << 16),                                    // data_size=2B
             (int)((td0 & 0xffffu) << 16),                       // tensor_dim0 lo16
             (int)(((td0 >> 16) & 0xffffu) | ((td1 & 0xffffu) << 16)),
             (int)(((td1 >> 16) & 0xffffu) | (32u << 16)),       // tile_dim0=32
             (int)32,                                            // tile_dim1=32
             (int)(st0 & 0xffffffffu),
             (int)((st0 >> 32) & 0xffffu),                       // stride1=0
             0 };
  v4i g2 = { 0, 0, 0, 0 };
  v4i g3 = { 0, 0, 0, 0 };
#if __clang_major__ >= 23
  v8i g4 = { 0, 0, 0, 0, 0, 0, 0, 0 };
  __builtin_amdgcn_tensor_load_to_lds(g0, g1, g2, g3, g4, 0);
#else
  __builtin_amdgcn_tensor_load_to_lds(g0, g1, g2, g3, 0);
#endif
}
#endif

// ---------------- generic utility kernels ----------------
__global__ void k_zero_f32(float* p, long n) {
  long i = blockIdx.x * (long)blockDim.x + threadIdx.x;
  if (i < n) p[i] = 0.0f;
}
__global__ void k_zero_f16(half_t* p, long n) {
  long i = blockIdx.x * (long)blockDim.x + threadIdx.x;
  if (i < n) p[i] = (half_t)0.0f;
}
__global__ void k_copy_f32(const float* __restrict__ in, float* __restrict__ out, long n) {
  long i = blockIdx.x * (long)blockDim.x + threadIdx.x;
  if (i < n) out[i] = in[i];
}
__global__ void k_cvt(const float* __restrict__ in, half_t* __restrict__ out, long n) {
  long i = blockIdx.x * (long)blockDim.x + threadIdx.x;
  if (i < n) out[i] = (half_t)in[i];
}
// out[r*ldo + c] = in[r*C + c]  (pad columns beyond C stay whatever was there / pre-zeroed)
__global__ void k_cvt_pad(const float* __restrict__ in, half_t* __restrict__ out,
                          int R, int C, int ldo) {
  long i = blockIdx.x * (long)blockDim.x + threadIdx.x;
  long total = (long)R * C;
  if (i >= total) return;
  int c = (int)(i % C), r = (int)(i / C);
  out[(long)r * ldo + c] = (half_t)in[i];
}
// out[c*R + r] = in[r*C + c]   (in: R x C f32, out: C x R f16)
__global__ void k_tcvt(const float* __restrict__ in, half_t* __restrict__ out, int R, int C) {
  long i = blockIdx.x * (long)blockDim.x + threadIdx.x;
  long total = (long)R * C;
  if (i >= total) return;
  int c = (int)(i % C), r = (int)(i / C);
  out[(long)c * R + r] = (half_t)in[i];
}

// ---------------- WMMA GEMM (guard-free; all dims full tiles by construction) ----------
// C[M,N] = act(A[M,K]f16 * B[K,N]f16 + bias).  M%32==0, K%32==0, N%64==0 at all call sites.
// 256 threads = 8 waves; block tile 32x64; wave grid 2x4 of 16x16 WMMA tiles; K step 32.
// Software-pipelined: while computing tile k, tile k+32 is staged — A by the Tensor Data
// Mover (tensor_load_to_lds / TENSORcnt) into the alternate LDS buffer, B by a b128 VMEM
// load held in registers and committed to LDS (transposed [n][k]) after the WMMA.
__global__ void k_gemm(const half_t* __restrict__ A, const half_t* __restrict__ Bm,
                       const float* __restrict__ bias, float* __restrict__ C,
                       int K, int lda, int ldb, int ldc, int act) {
  __shared__ __align__(16) half_t As[2][32 * 32];   // row-major [m][k]
  __shared__ __align__(16) half_t Bs[2][64 * 32];   // transposed [n][k]
  int tid = threadIdx.x;
  int m0 = blockIdx.y * 32, n0 = blockIdx.x * 64;
  int ar = tid >> 3, ac = (tid & 7) * 4;            // A fallback coop copy (b64/thread)
  int bk = tid >> 3, bn = (tid & 7) * 8;            // B coop load (b128/thread)
  int wave = tid >> 5, lane = tid & 31;
  int wm = wave >> 2, wn = wave & 3;
  int g = lane >> 4, lr = lane & 15;
  (void)ar; (void)ac;
  const half_t* Arow = A + (long)m0 * lda;
  v8f acc = {};
  // prologue: stage tile k0=0 into buffer 0
  {
    v8h t0 = *(const v8h*)(Bm + (long)bk * ldb + n0 + bn);
#if HAVE_TDM
    if (wave == 0) tdm_load_a(Arow, &As[0][0], lda);
#else
    *(uint2*)(&As[0][ar * 32 + ac]) = *(const uint2*)(Arow + (long)ar * lda + ac);
#endif
#pragma unroll
    for (int i = 0; i < 8; i++) Bs[0][(bn + i) * 32 + bk] = t0[i];
#if HAVE_TDM
    if (wave == 0) __builtin_amdgcn_s_wait_tensorcnt(0);
#endif
    __syncthreads();
  }
  for (int k0 = 0; k0 < K; k0 += 32) {
    int cur = (k0 >> 5) & 1, nxt = cur ^ 1;
    bool has_next = (k0 + 32) < K;
    v8h t = {};
    if (has_next) {                                  // stage k0+32 (overlaps compute)
      t = *(const v8h*)(Bm + (long)(k0 + 32 + bk) * ldb + n0 + bn);
#if HAVE_TDM
      if (wave == 0) tdm_load_a(Arow + (k0 + 32), &As[nxt][0], lda);
#else
      *(uint2*)(&As[nxt][ar * 32 + ac]) =
          *(const uint2*)(Arow + (long)ar * lda + (k0 + 32) + ac);
#endif
      if (k0 + 64 < K)
        __builtin_prefetch(Bm + (long)(k0 + 64 + bk) * ldb + n0 + bn, 0, 1);
    }
    // fragment gather per ISA 16-bit layout: lane l(row m=l&15), halves 0..7 ->
    // K = 8*(l>>4)+0..7, halves 8..15 -> K = 16+8*(l>>4)+0..7; Bs is [n][k] so same gather.
    v8h alo = *(const v8h*)(&As[cur][(wm * 16 + lr) * 32 + 8 * g]);
    v8h ahi = *(const v8h*)(&As[cur][(wm * 16 + lr) * 32 + 16 + 8 * g]);
    v8h blo = *(const v8h*)(&Bs[cur][(wn * 16 + lr) * 32 + 8 * g]);
    v8h bhi = *(const v8h*)(&Bs[cur][(wn * 16 + lr) * 32 + 16 + 8 * g]);
    v16h a, b;
#pragma unroll
    for (int i = 0; i < 8; i++) { a[i] = alo[i]; a[8 + i] = ahi[i]; b[i] = blo[i]; b[8 + i] = bhi[i]; }
    acc = __builtin_amdgcn_wmma_f32_16x16x32_f16(false, a, false, b, (short)0, acc, false, false);
    if (has_next) {
#pragma unroll
      for (int i = 0; i < 8; i++) Bs[nxt][(bn + i) * 32 + bk] = t[i];
#if HAVE_TDM
      if (wave == 0) __builtin_amdgcn_s_wait_tensorcnt(0);
#endif
    }
    __syncthreads();
  }
  int row = m0 + wm * 16 + g * 8;
  int col = n0 + wn * 16 + lr;
  float bv = bias ? bias[col] : 0.0f;
#pragma unroll
  for (int i = 0; i < 8; i++) {
    float v = acc[i] + bv;
    if (act == 1) v = fmaxf(v, 0.0f);
    C[(long)(row + i) * ldc + col] = v;
  }
}

// ---------------- GAT1 ----------------
__global__ void k_gat1(const float* __restrict__ x, const float* Wl, const float* bl,
                       const float* Wr, const float* br, float* xl, float* xr, long total) {
  long i = blockIdx.x * (long)blockDim.x + threadIdx.x;
  if (i >= total) return;
  long row = i >> 9; int j = (int)(i & 511);
  float x0 = x[row * 2 + 0], x1 = x[row * 2 + 1];
  xl[i] = x0 * Wl[j] + x1 * Wl[512 + j] + bl[j];
  xr[i] = x0 * Wr[j] + x1 * Wr[512 + j] + br[j];
}
__global__ void k_logits1(const float* __restrict__ xl, const float* __restrict__ xr,
                          const int* __restrict__ ei, const float* __restrict__ att,
                          float* logit, unsigned* mkey, long total) {
  long i = blockIdx.x * (long)blockDim.x + threadIdx.x;
  if (i >= total) return;
  int h = (int)(i & 3); long ge = i >> 2;
  int e = (int)(ge % ET), gg = (int)(ge / ET);
  int s, d; edge_nodes(ei, e, s, d);
  const float* pl = xl + ((long)gg * Nn + s) * 512 + h * 128;
  const float* pr = xr + ((long)gg * Nn + d) * 512 + h * 128;
  const float* pa = att + h * 128;
  float acc = 0.0f;
  for (int c = 0; c < 128; c++) {
    float v = pl[c] + pr[c];
    v = v > 0.0f ? v : v * NEG_SLOPE;
    acc += v * pa[c];
  }
  logit[i] = acc;
  atomicMax(&mkey[((long)gg * Nn + d) * 4 + h], fkey(acc));
}
__global__ void k_soft1(const unsigned* __restrict__ mkey, const int* __restrict__ ei,
                        float* logit, float* den, long total) {
  long i = blockIdx.x * (long)blockDim.x + threadIdx.x;
  if (i >= total) return;
  int h = (int)(i & 3); long ge = i >> 2;
  int e = (int)(ge % ET), gg = (int)(ge / ET);
  int s, d; edge_nodes(ei, e, s, d);
  long nd = (long)gg * Nn + d;
  float a = expf(logit[i] - fromkey(mkey[nd * 4 + h]));
  logit[i] = a;
  atomicAdd(&den[nd * 4 + h], a);
}
__global__ void k_agg1(const float* __restrict__ xl, const float* __restrict__ logit,
                       const float* __restrict__ den, const int* __restrict__ ei,
                       float* out, long total) {
  long i = blockIdx.x * (long)blockDim.x + threadIdx.x;
  if (i >= total) return;
  int j = (int)(i & 511); long ge = i >> 9;
  int e = (int)(ge % ET), gg = (int)(ge / ET);
  int s, d; edge_nodes(ei, e, s, d);
  int h = j >> 7;
  long nd = (long)gg * Nn + d, ns = (long)gg * Nn + s;
  float alpha = logit[ge * 4 + h] / den[nd * 4 + h];
  atomicAdd(&out[nd * 512 + j], alpha * xl[ns * 512 + j]);
}
__global__ void k_fin1(const float* __restrict__ out1, const float* __restrict__ bias,
                       half_t* s1h, long total) {
  long i = blockIdx.x * (long)blockDim.x + threadIdx.x;
  if (i >= total) return;
  int j = (int)(i & 511);
  float v = out1[i] + bias[j];
  v = v > 0.0f ? v : expf(v) - 1.0f;         // ELU
  s1h[i] = (half_t)v;
}

// ---------------- GAT2 (H=1, C=32; xl2/xr2 have stride 64) ----------------
__global__ void k_logits2(const float* __restrict__ xl, const float* __restrict__ xr,
                          const int* __restrict__ ei, const float* __restrict__ att,
                          float* logit, unsigned* mkey, long total) {
  long i = blockIdx.x * (long)blockDim.x + threadIdx.x;
  if (i >= total) return;
  int e = (int)(i % ET), gg = (int)(i / ET);
  int s, d; edge_nodes(ei, e, s, d);
  const float* pl = xl + ((long)gg * Nn + s) * 64;
  const float* pr = xr + ((long)gg * Nn + d) * 64;
  float acc = 0.0f;
  for (int c = 0; c < 32; c++) {
    float v = pl[c] + pr[c];
    v = v > 0.0f ? v : v * NEG_SLOPE;
    acc += v * att[c];
  }
  logit[i] = acc;
  atomicMax(&mkey[(long)gg * Nn + d], fkey(acc));
}
__global__ void k_soft2(const unsigned* __restrict__ mkey, const int* __restrict__ ei,
                        float* logit, float* den, long total) {
  long i = blockIdx.x * (long)blockDim.x + threadIdx.x;
  if (i >= total) return;
  int e = (int)(i % ET), gg = (int)(i / ET);
  int s, d; edge_nodes(ei, e, s, d);
  long nd = (long)gg * Nn + d;
  float a = expf(logit[i] - fromkey(mkey[nd]));
  logit[i] = a;
  atomicAdd(&den[nd], a);
}
__global__ void k_agg2(const float* __restrict__ xl, const float* __restrict__ logit,
                       const float* __restrict__ den, const int* __restrict__ ei,
                       float* out, long total) {
  long i = blockIdx.x * (long)blockDim.x + threadIdx.x;
  if (i >= total) return;
  int c = (int)(i & 31); long ge = i >> 5;
  int e = (int)(ge % ET), gg = (int)(ge / ET);
  int s, d; edge_nodes(ei, e, s, d);
  long nd = (long)gg * Nn + d, ns = (long)gg * Nn + s;
  float alpha = logit[ge] / den[nd];
  atomicAdd(&out[nd * 32 + c], alpha * xl[ns * 64 + c]);
}
// LayerNorm(32) + reorder (g=b*T+t, n) -> time-major LSTM input (t, b*N+n) f16
__global__ void k_ln(const float* __restrict__ out2, const float* __restrict__ bias,
                     const float* __restrict__ gam, const float* __restrict__ bet,
                     half_t* inh) {
  long row = blockIdx.x * 8 + (threadIdx.x >> 5);
  int c = threadIdx.x & 31;
  if (row >= NN) return;
  float v = out2[row * 32 + c] + bias[c];
  float s = v;
  for (int o = 16; o > 0; o >>= 1) s += __shfl_xor(s, o, 32);
  float mu = s * (1.0f / 32.0f);
  float dv = v - mu;
  float q = dv * dv;
  for (int o = 16; o > 0; o >>= 1) q += __shfl_xor(q, o, 32);
  float y = dv * rsqrtf(q * (1.0f / 32.0f) + EPSc) * gam[c] + bet[c];
  int gg = (int)(row / Nn), n = (int)(row % Nn);
  int b = gg / T_INc, t = gg % T_INc;
  long bn = (long)b * Nn + n;
  inh[((long)t * BNr + bn) * 32 + c] = (half_t)y;
}

// ---------------- LSTM cell ----------------
__global__ void k_cell(const float* __restrict__ gx, const float* __restrict__ gh,
                       const float* __restrict__ bhh, float* __restrict__ seqout,
                       half_t* __restrict__ hh, float* __restrict__ cbuf, int t, int dofs) {
  long i = blockIdx.x * (long)blockDim.x + threadIdx.x;
  if (i >= (long)BNr * 128) return;
  int bn = (int)(i >> 7), j = (int)(i & 127);
  const float* gxr = gx + ((long)t * BNr + bn) * 512;
  const float* ghr = gh + (long)bn * 512;
  float gi = gxr[j]       + ghr[j]       + bhh[j];
  float gf = gxr[128 + j] + ghr[128 + j] + bhh[128 + j];
  float gG = gxr[256 + j] + ghr[256 + j] + bhh[256 + j];
  float go = gxr[384 + j] + ghr[384 + j] + bhh[384 + j];
  float c = cbuf[i];
  c = sigm(gf) * c + sigm(gi) * tanhf(gG);
  float h = sigm(go) * tanhf(c);
  cbuf[i] = c;
  hh[i] = (half_t)h;
  seqout[((long)t * BNr + bn) * 256 + dofs + j] = h;
}

// ---------------- head scatter: out[b][t][n] = outh[b*N+n][t] (outh stride 64) ----------
__global__ void k_scatter(const float* __restrict__ outh, float* __restrict__ out) {
  long i = blockIdx.x * (long)blockDim.x + threadIdx.x;
  long total = (long)Bc * T_OUTc * Nn;
  if (i >= total) return;
  int n = (int)(i % Nn);
  int t = (int)((i / Nn) % T_OUTc);
  int b = (int)(i / ((long)Nn * T_OUTc));
  out[i] = outh[((long)b * Nn + n) * 64 + t];
}

// ---------------- host orchestration ----------------
static inline unsigned gsz(long n) { return (unsigned)((n + 255) / 256); }

extern "C" void kernel_launch(void* const* d_in, const int* in_sizes, int n_in,
                              void* d_out, int out_size, void* d_ws, size_t ws_size,
                              hipStream_t stream) {
  (void)in_sizes; (void)n_in; (void)out_size; (void)ws_size;
  const float* x      = (const float*)d_in[0];
  const int*   ei     = (const int*)d_in[1];
  const float* g1Wl   = (const float*)d_in[2];
  const float* g1bl   = (const float*)d_in[3];
  const float* g1Wr   = (const float*)d_in[4];
  const float* g1br   = (const float*)d_in[5];
  const float* g1att  = (const float*)d_in[6];
  const float* g1bias = (const float*)d_in[7];
  const float* g2Wl   = (const float*)d_in[8];
  const float* g2bl   = (const float*)d_in[9];
  const float* g2Wr   = (const float*)d_in[10];
  const float* g2br   = (const float*)d_in[11];
  const float* g2att  = (const float*)d_in[12];
  const float* g2bias = (const float*)d_in[13];
  const float* lnG    = (const float*)d_in[14];
  const float* lnB    = (const float*)d_in[15];
  const float* hW1    = (const float*)d_in[16];
  const float* hb1    = (const float*)d_in[17];
  const float* hW2    = (const float*)d_in[18];
  const float* hb2    = (const float*)d_in[19];

  char* ws = (char*)d_ws;
  float*    XL1    = (float*)(ws + OFF_A);
  float*    GATESX = (float*)(ws + OFF_A);
  float*    XR1    = (float*)(ws + OFF_B);
  half_t*   S1H    = (half_t*)(ws + B_S1H);
  half_t*   INH    = (half_t*)(ws + B_S1H);
  float*    SEQOUT = (float*)(ws + B_SEQ);
  float*    OUT1   = (float*)(ws + OFF_C);
  float*    XL2    = (float*)(ws + C_XL2);
  float*    XR2    = (float*)(ws + C_XR2);
  float*    OUT2   = (float*)(ws + C_OUT2);
  float*    LOG2   = (float*)(ws + C_LOG2);
  unsigned* M2     = (unsigned*)(ws + C_M2);
  float*    DEN2   = (float*)(ws + C_DEN2);
  float*    H1     = (float*)(ws + C_H1);
  half_t*   H1H    = (half_t*)(ws + C_H1H);
  float*    OUTH   = (float*)(ws + C_OUTH);
  float*    LOG1   = (float*)(ws + D_LOG1);
  unsigned* M1     = (unsigned*)(ws + D_M1);
  float*    DEN1   = (float*)(ws + D_DEN1);
  half_t*   W1H    = (half_t*)(ws + D_W1H);
  half_t*   W2H    = (half_t*)(ws + D_W2H);
  half_t*   WL2H   = (half_t*)(ws + D_WL2H);
  half_t*   WR2H   = (half_t*)(ws + D_WR2H);
  half_t*   WIHT   = (half_t*)(ws + D_WIHT);   // slots of 131072 halfs
  half_t*   WHHT   = (half_t*)(ws + D_WHHT);   // slots of 65536 halfs
  float*    BL2P   = (float*)(ws + D_BL2P);
  float*    BR2P   = (float*)(ws + D_BR2P);
  float*    B2P    = (float*)(ws + D_B2P);
  float*    GATESH = (float*)(ws + D_GATESH);
  half_t*   HH     = (half_t*)(ws + D_HH);
  float*    CBUF   = (float*)(ws + D_CBUF);
  half_t*   CTXH   = (half_t*)(ws + D_CTXH);

  auto gemm = [&](const half_t* A, const half_t* Bm, const float* bias, float* C,
                  int M, int N, int K, int lda, int ldb, int ldc, int act) {
    dim3 grid((unsigned)(N / 64), (unsigned)(M / 32));
    k_gemm<<<grid, 256, 0, stream>>>(A, Bm, bias, C, K, lda, ldb, ldc, act);
  };

  // ---- weight prep (persistent f16 copies, N padded to multiples of 64) ----
  for (int l = 0; l < 2; l++)
    for (int d = 0; d < 2; d++) {
      int din = (l == 0) ? 32 : 256;
      const float* Wih = (const float*)d_in[20 + l * 8 + d * 4 + 0];
      const float* Whh = (const float*)d_in[20 + l * 8 + d * 4 + 1];
      k_tcvt<<<gsz((long)512 * din), 256, 0, stream>>>(Wih, WIHT + (l * 2 + d) * 131072, 512, din);
      k_tcvt<<<gsz((long)512 * 128), 256, 0, stream>>>(Whh, WHHT + (l * 2 + d) * 65536, 512, 128);
    }
  k_cvt<<<gsz(256 * 128), 256, 0, stream>>>(hW1, W1H, 256 * 128);
  k_zero_f16<<<gsz(128 * 64), 256, 0, stream>>>(W2H, 128 * 64);
  k_cvt_pad<<<gsz(128 * 12), 256, 0, stream>>>(hW2, W2H, 128, 12, 64);
  k_zero_f16<<<gsz(512 * 64), 256, 0, stream>>>(WL2H, 512 * 64);
  k_cvt_pad<<<gsz(512 * 32), 256, 0, stream>>>(g2Wl, WL2H, 512, 32, 64);
  k_zero_f16<<<gsz(512 * 64), 256, 0, stream>>>(WR2H, 512 * 64);
  k_cvt_pad<<<gsz(512 * 32), 256, 0, stream>>>(g2Wr, WR2H, 512, 32, 64);
  k_zero_f32<<<1, 256, 0, stream>>>(BL2P, 64);
  k_copy_f32<<<1, 256, 0, stream>>>(g2bl, BL2P, 32);
  k_zero_f32<<<1, 256, 0, stream>>>(BR2P, 64);
  k_copy_f32<<<1, 256, 0, stream>>>(g2br, BR2P, 32);
  k_zero_f32<<<1, 256, 0, stream>>>(B2P, 64);
  k_copy_f32<<<1, 256, 0, stream>>>(hb2, B2P, 12);

  // ---- GAT layer 1 ----
  k_gat1<<<gsz(NN * 512), 256, 0, stream>>>(x, g1Wl, g1bl, g1Wr, g1br, XL1, XR1, NN * 512);
  k_zero_f32<<<gsz(NN * 4), 256, 0, stream>>>((float*)M1, NN * 4);
  k_zero_f32<<<gsz(NN * 4), 256, 0, stream>>>(DEN1, NN * 4);
  k_zero_f32<<<gsz(NN * 512), 256, 0, stream>>>(OUT1, NN * 512);
  long tE1 = (long)NG * ET * 4;
  k_logits1<<<gsz(tE1), 256, 0, stream>>>(XL1, XR1, ei, g1att, LOG1, M1, tE1);
  k_soft1<<<gsz(tE1), 256, 0, stream>>>(M1, ei, LOG1, DEN1, tE1);
  long tA1 = (long)NG * ET * 512;
  k_agg1<<<gsz(tA1), 256, 0, stream>>>(XL1, LOG1, DEN1, ei, OUT1, tA1);
  k_fin1<<<gsz(NN * 512), 256, 0, stream>>>(OUT1, g1bias, S1H, NN * 512);

  // ---- GAT layer 2 projections via WMMA (N padded 32->64) ----
  gemm(S1H, WL2H, BL2P, XL2, (int)NN, 64, 512, 512, 64, 64, 0);
  gemm(S1H, WR2H, BR2P, XR2, (int)NN, 64, 512, 512, 64, 64, 0);
  k_zero_f32<<<gsz(NN), 256, 0, stream>>>((float*)M2, NN);
  k_zero_f32<<<gsz(NN), 256, 0, stream>>>(DEN2, NN);
  k_zero_f32<<<gsz(NN * 32), 256, 0, stream>>>(OUT2, NN * 32);
  long tE2 = (long)NG * ET;
  k_logits2<<<gsz(tE2), 256, 0, stream>>>(XL2, XR2, ei, g2att, LOG2, M2, tE2);
  k_soft2<<<gsz(tE2), 256, 0, stream>>>(M2, ei, LOG2, DEN2, tE2);
  long tA2 = (long)NG * ET * 32;
  k_agg2<<<gsz(tA2), 256, 0, stream>>>(XL2, LOG2, DEN2, ei, OUT2, tA2);
  k_ln<<<(unsigned)(NN / 8), 256, 0, stream>>>(OUT2, g2bias, lnG, lnB, INH);

  // ---- BiLSTM (2 layers) ----
  for (int l = 0; l < 2; l++) {
    int din = (l == 0) ? 32 : 256;
    for (int d = 0; d < 2; d++) {
      const half_t* WT = WIHT + (l * 2 + d) * 131072;
      const half_t* HT = WHHT + (l * 2 + d) * 65536;
      const float* bih = (const float*)d_in[20 + l * 8 + d * 4 + 2];
      const float* bhh = (const float*)d_in[20 + l * 8 + d * 4 + 3];
      gemm(INH, WT, bih, GATESX, (int)NN, 512, din, din, 512, 512, 0);   // all timesteps
      k_zero_f16<<<gsz((long)BNr * 128), 256, 0, stream>>>(HH, (long)BNr * 128);
      k_zero_f32<<<gsz((long)BNr * 128), 256, 0, stream>>>(CBUF, (long)BNr * 128);
      for (int s = 0; s < T_INc; s++) {
        int t = (d == 0) ? s : (T_INc - 1 - s);
        gemm(HH, HT, nullptr, GATESH, BNr, 512, 128, 128, 512, 512, 0);  // recurrent
        k_cell<<<gsz((long)BNr * 128), 256, 0, stream>>>(GATESX, GATESH, bhh, SEQOUT, HH,
                                                         CBUF, t, d * 128);
      }
    }
    if (l == 0) k_cvt<<<gsz(NN * 256), 256, 0, stream>>>(SEQOUT, INH, NN * 256);
  }

  // ---- head MLP via WMMA (head2 N padded 12->64) ----
  k_cvt<<<gsz((long)BNr * 256), 256, 0, stream>>>(SEQOUT + (long)(T_INc - 1) * BNr * 256, CTXH,
                                                  (long)BNr * 256);
  gemm(CTXH, W1H, hb1, H1, BNr, 128, 256, 256, 128, 128, 1);  // ReLU
  k_cvt<<<gsz((long)BNr * 128), 256, 0, stream>>>(H1, H1H, (long)BNr * 128);
  gemm(H1H, W2H, B2P, OUTH, BNr, 64, 128, 128, 64, 64, 0);
  k_scatter<<<gsz((long)Bc * T_OUTc * Nn), 256, 0, stream>>>(OUTH, (float*)d_out);
}
